// GIN_34531537060539
// MI455X (gfx1250) — matrix operand order repaired and checked by reading, not compiled, not run on hardware
//
#include <hip/hip_runtime.h>
#include <hip/hip_bf16.h>

// ---------------------------------------------------------------------------
// GIN forward for MI455X (gfx1250, wave32).
// GEMMs run on v_wmma_f32_16x16x32_bf16 (bf16 A/B, f32 accum), 32x64 tile per
// wave (2x4 WMMA grid), branch-free inner loop (rows clamped, not masked).
// Segment sums (edge agg, graph pooling) use global f32 atomics (L2-native).
// ---------------------------------------------------------------------------

#define Nn   50000
#define Ee   800000
#define Gg   64
#define INF_ 256
#define HID_ 512
#define OUT_ 10

typedef unsigned short u16;
typedef __attribute__((ext_vector_type(16))) __bf16 v16bf;
typedef __attribute__((ext_vector_type(8)))  float  v8f;

union Frag { v16bf v; uint4 q[2]; };

__device__ __forceinline__ u16 f32_to_bf16(float f) {
    unsigned int u = __float_as_uint(f);
    if ((u & 0x7fffffffu) > 0x7f800000u) return (u16)((u >> 16) | 0x0040u); // quiet NaN
    u += 0x7fffu + ((u >> 16) & 1u);                                        // RNE
    return (u16)(u >> 16);
}

// ----------------------------- utility kernels -----------------------------

__global__ void k_zero_f32(float* __restrict__ p, int n) {
    int i = blockIdx.x * blockDim.x + threadIdx.x;
    if (i < n) p[i] = 0.f;
}

__global__ void k_zero_u16(u16* __restrict__ p, int n) {
    int i = blockIdx.x * blockDim.x + threadIdx.x;
    if (i < n) p[i] = 0;
}

__global__ void k_add_inplace(float* __restrict__ a, const float* __restrict__ b, int n) {
    int i = blockIdx.x * blockDim.x + threadIdx.x;
    if (i < n) a[i] += b[i];
}

__global__ void k_to_bf16(const float* __restrict__ x, u16* __restrict__ y, int n) {
    int i = blockIdx.x * blockDim.x + threadIdx.x;
    if (i < n) y[i] = f32_to_bf16(x[i]);
}

// W [K x Nc] f32 row-major  ->  Wt [Nc x K] bf16 row-major
__global__ void k_transpose_bf16(const float* __restrict__ W, u16* __restrict__ Wt,
                                 int K, int Nc) {
    int i = blockIdx.x * blockDim.x + threadIdx.x;
    if (i >= K * Nc) return;
    int k = i / Nc, n = i % Nc;
    Wt[n * K + k] = f32_to_bf16(W[i]);
}

// agg[dst] += h[src] ; one thread per (edge, 4-float chunk)
__global__ void k_edge_agg(const float* __restrict__ h, const int* __restrict__ src,
                           const int* __restrict__ dst, float* __restrict__ agg,
                           int E, int d) {
    int chunks = d >> 2;
    int idx = blockIdx.x * blockDim.x + threadIdx.x;
    if (idx >= E * chunks) return;
    int e = idx / chunks;
    int c = (idx - e * chunks) << 2;
    int s = src[e], t = dst[e];
    const float4 v = *(const float4*)(h + (size_t)s * d + c);
    float* o = agg + (size_t)t * d + c;
    atomicAdd(o + 0, v.x); atomicAdd(o + 1, v.y);
    atomicAdd(o + 2, v.z); atomicAdd(o + 3, v.w);
}

// ------------------------------- WMMA GEMM ---------------------------------
// C[M x Nc] = A[M x K] * B[K x Nc], A bf16 row-major, B given as Bt[Nc x K] bf16.
// 128 threads = 4 waves; block tile 64(M) x 128(N); each wave 32x64 (2x4 WMMA).
// Out-of-range M rows are CLAMPED (duplicate finite data) so the K-loop is
// branch-free; their results are simply never stored.
__global__ __launch_bounds__(128)
void k_gemm_bf16(const u16* __restrict__ A, const u16* __restrict__ Bt,
                 float* __restrict__ C, int M, int K, int Nc) {
    const int lane = threadIdx.x & 31;
    const int wave = threadIdx.x >> 5;
    const int wr = wave >> 1, wc = wave & 1;
    const int l15  = lane & 15;
    const int half = lane >> 4;
    const int mBase = blockIdx.x * 64 + wr * 32;
    const int nBase = blockIdx.y * 128 + wc * 64;

    v8f acc[2][4] = {};

    // A fragment per lane: K in {k0..k0+7} U {k0+16..k0+23}, k0 = kb + half*8
    int rowA0 = mBase + l15;       if (rowA0 >= M) rowA0 = M - 1;
    int rowA1 = mBase + 16 + l15;  if (rowA1 >= M) rowA1 = M - 1;
    const u16* ap0 = A + (size_t)rowA0 * K + half * 8;
    const u16* ap1 = A + (size_t)rowA1 * K + half * 8;

    // B fragment per lane: K in {k0..k0+7} U {k0+16..k0+23}, k0 = kb + half*8 (Bt row = output col)
    const u16* bp[4];
    #pragma unroll
    for (int tj = 0; tj < 4; ++tj)
        bp[tj] = Bt + (size_t)(nBase + tj * 16 + l15) * K + half * 8;

    for (int kb = 0; kb < K; kb += 32) {
        __builtin_prefetch(ap0 + kb + 64, 0, 1);   // global_prefetch_b8
        __builtin_prefetch(ap1 + kb + 64, 0, 1);

        Frag a[2], b[4];
        a[0].q[0] = *(const uint4*)(ap0 + kb);
        a[0].q[1] = *(const uint4*)(ap0 + kb + 16);
        a[1].q[0] = *(const uint4*)(ap1 + kb);
        a[1].q[1] = *(const uint4*)(ap1 + kb + 16);
        #pragma unroll
        for (int tj = 0; tj < 4; ++tj) {
            b[tj].q[0] = *(const uint4*)(bp[tj] + kb);
            b[tj].q[1] = *(const uint4*)(bp[tj] + kb + 16);
        }

        // (neg_a, A, neg_b, B, c_mod, C, reuse_a, reuse_b)
        #pragma unroll
        for (int ti = 0; ti < 2; ++ti)
            #pragma unroll
            for (int tj = 0; tj < 4; ++tj)
                acc[ti][tj] = __builtin_amdgcn_wmma_f32_16x16x32_bf16(
                    false, a[ti].v, false, b[tj].v, (short)0, acc[ti][tj], false, false);
    }

    // C/D layout: VGPR r -> row (r + 8*half), col = l15
    #pragma unroll
    for (int ti = 0; ti < 2; ++ti) {
        #pragma unroll
        for (int tj = 0; tj < 4; ++tj) {
            int col = nBase + tj * 16 + l15;
            if (col >= Nc) continue;
            #pragma unroll
            for (int r = 0; r < 8; ++r) {
                int row = mBase + ti * 16 + half * 8 + r;
                if (row < M) C[(size_t)row * Nc + col] = acc[ti][tj][r];
            }
        }
    }
}

// ---------------------------- batch-norm pieces -----------------------------

// Partial column sums / sums-of-squares; coalesced rows, few atomics.
__global__ void k_col_stats(const float* __restrict__ t, float* __restrict__ csum,
                            float* __restrict__ csq, int M, int d, int rowsPerBlock) {
    int r0 = blockIdx.x * rowsPerBlock;
    int r1 = min(M, r0 + rowsPerBlock);
    for (int c = threadIdx.x; c < d; c += blockDim.x) {
        float s = 0.f, q = 0.f;
        for (int r = r0; r < r1; ++r) {
            float v = t[(size_t)r * d + c];
            s += v; q += v * v;
        }
        atomicAdd(&csum[c], s);
        atomicAdd(&csq[c],  q);
    }
}

__global__ void k_finalize_stats(const float* __restrict__ csum, const float* __restrict__ csq,
                                 float* __restrict__ mean, float* __restrict__ rsig,
                                 int M, int d) {
    int c = blockIdx.x * blockDim.x + threadIdx.x;
    if (c >= d) return;
    float mu  = csum[c] / (float)M;
    float var = fmaxf(csq[c] / (float)M - mu * mu, 0.f);
    mean[c] = mu;
    rsig[c] = rsqrtf(var + 1e-5f);
}

__global__ void k_bn_relu_bf16(const float* __restrict__ x, const float* __restrict__ mean,
                               const float* __restrict__ rsig, const float* __restrict__ g,
                               const float* __restrict__ b, u16* __restrict__ y,
                               int n, int d) {
    int i = blockIdx.x * blockDim.x + threadIdx.x;
    if (i >= n) return;
    int c = i % d;
    float v = fmaf(g[c] * rsig[c], x[i] - mean[c], b[c]);
    y[i] = f32_to_bf16(fmaxf(v, 0.f));
}

__global__ void k_bn_relu_f32(const float* __restrict__ x, const float* __restrict__ mean,
                              const float* __restrict__ rsig, const float* __restrict__ g,
                              const float* __restrict__ b, float* __restrict__ y,
                              int n, int d) {
    int i = blockIdx.x * blockDim.x + threadIdx.x;
    if (i >= n) return;
    int c = i % d;
    float v = fmaf(g[c] * rsig[c], x[i] - mean[c], b[c]);
    y[i] = fmaxf(v, 0.f);
}

// -------------------------- pooling / heads / softmax -----------------------

__global__ void k_pool(const float* __restrict__ h, const int* __restrict__ gid,
                       float* __restrict__ pooled, int N, int d) {
    int i = blockIdx.x * blockDim.x + threadIdx.x;
    if (i >= N * d) return;
    int r = i / d, c = i - r * d;
    atomicAdd(&pooled[gid[r] * d + c], h[i]);
}

// score[g][o] += pooled[g] . W[:,o] + b[o]   (640 threads, 1 block)
__global__ void k_score(const float* __restrict__ pooled, const float* __restrict__ W,
                        const float* __restrict__ bias, float* __restrict__ score, int d) {
    int tid = threadIdx.x;
    if (tid >= Gg * OUT_) return;
    int g = tid / OUT_, o = tid - g * OUT_;
    float s = bias[o];
    for (int c = 0; c < d; ++c) s = fmaf(pooled[g * d + c], W[c * OUT_ + o], s);
    score[tid] += s;
}

__global__ void k_logsoftmax(const float* __restrict__ score, float* __restrict__ out) {
    int g = threadIdx.x;
    if (g >= Gg) return;
    float mx = -3.402823466e38f;
    for (int o = 0; o < OUT_; ++o) mx = fmaxf(mx, score[g * OUT_ + o]);
    float se = 0.f;
    for (int o = 0; o < OUT_; ++o) se += __expf(score[g * OUT_ + o] - mx);
    float lse = mx + __logf(se);
    for (int o = 0; o < OUT_; ++o) out[g * OUT_ + o] = score[g * OUT_ + o] - lse;
}

// --------------------------------- driver -----------------------------------

static inline int cdiv(int a, int b) { return (a + b - 1) / b; }

extern "C" void kernel_launch(void* const* d_in, const int* in_sizes, int n_in,
                              void* d_out, int out_size, void* d_ws, size_t ws_size,
                              hipStream_t stream) {
    (void)in_sizes; (void)n_in; (void)out_size; (void)ws_size;

    const float* feat = (const float*)d_in[0];
    const int*   src  = (const int*)d_in[51];
    const int*   dst  = (const int*)d_in[52];
    const int*   gid  = (const int*)d_in[53];

    // ---- workspace carve-out (≈257 MB) ----
    size_t off = 0;
    auto carve = [&](size_t bytes) -> void* {
        void* p = (char*)d_ws + off;
        off += (bytes + 255) & ~(size_t)255;
        return p;
    };
    float* h      = (float*)carve((size_t)Nn * HID_ * sizeof(float));  // hidden
    float* m      = (float*)carve((size_t)Nn * HID_ * sizeof(float));  // agg / m / GEMM out (t)
    u16*   abf    = (u16*)  carve((size_t)Nn * HID_ * sizeof(u16));    // bf16 activations
    u16*   wbf    = (u16*)  carve((size_t)HID_ * HID_ * sizeof(u16));  // bf16 transposed weight
    float* mean   = (float*)carve(HID_ * sizeof(float));
    float* rsig   = (float*)carve(HID_ * sizeof(float));
    float* csum   = (float*)carve(HID_ * sizeof(float));
    float* csq    = (float*)carve(HID_ * sizeof(float));
    float* pooled = (float*)carve((size_t)Gg * HID_ * sizeof(float));
    float* score  = (float*)carve((size_t)Gg * OUT_ * sizeof(float));
    float* t      = m;   // GEMM output aliases m (GEMM reads only abf/wbf)

    const int B = 256;

    // score = 0 ; head 0 on raw features
    k_zero_f32<<<cdiv(Gg * OUT_, B), B, 0, stream>>>(score, Gg * OUT_);
    k_zero_f32<<<cdiv(Gg * INF_, B), B, 0, stream>>>(pooled, Gg * INF_);
    k_pool<<<cdiv(Nn * INF_, B), B, 0, stream>>>(feat, gid, pooled, Nn, INF_);
    k_score<<<1, Gg * OUT_, 0, stream>>>(pooled, (const float*)d_in[37],
                                         (const float*)d_in[44], score, INF_);

    // h = feat
    hipMemcpyAsync(h, feat, (size_t)Nn * INF_ * sizeof(float),
                   hipMemcpyDeviceToDevice, stream);

    const int din_arr[6]  = {INF_, HID_, HID_, HID_, HID_, HID_};
    const int dout_arr[6] = {HID_, HID_, HID_, HID_, HID_, 1};

    for (int i = 0; i < 6; ++i) {
        const int di = din_arr[i], dh = HID_, dp = dout_arr[i];
        const float* W1 = (const float*)d_in[1 + i * 6 + 0];
        const float* W2 = (const float*)d_in[1 + i * 6 + 1];
        const float* gm = (const float*)d_in[1 + i * 6 + 2];
        const float* bm = (const float*)d_in[1 + i * 6 + 3];
        const float* go = (const float*)d_in[1 + i * 6 + 4];
        const float* bo = (const float*)d_in[1 + i * 6 + 5];

        // ---- m = h + sum_{u->v} h_u ----
        k_zero_f32<<<cdiv(Nn * di, B), B, 0, stream>>>(m, Nn * di);
        k_edge_agg<<<cdiv(Ee * (di / 4), B), B, 0, stream>>>(h, src, dst, m, Ee, di);
        k_add_inplace<<<cdiv(Nn * di, B), B, 0, stream>>>(m, h, Nn * di);

        // ---- t = m @ W1 (bf16 WMMA) ----
        k_to_bf16<<<cdiv(Nn * di, B), B, 0, stream>>>(m, abf, Nn * di);
        k_transpose_bf16<<<cdiv(di * dh, B), B, 0, stream>>>(W1, wbf, di, dh);
        {
            dim3 grid(cdiv(Nn, 64), cdiv(dh, 128));
            k_gemm_bf16<<<grid, 128, 0, stream>>>(abf, wbf, t, Nn, di, dh);
        }

        // ---- h1 = relu(BN(t))  (bf16 for next GEMM) ----
        k_zero_f32<<<cdiv(dh, B), B, 0, stream>>>(csum, dh);
        k_zero_f32<<<cdiv(dh, B), B, 0, stream>>>(csq, dh);
        k_col_stats<<<cdiv(Nn, 128), B, 0, stream>>>(t, csum, csq, Nn, dh, 128);
        k_finalize_stats<<<cdiv(dh, B), B, 0, stream>>>(csum, csq, mean, rsig, Nn, dh);
        k_bn_relu_bf16<<<cdiv(Nn * dh, B), B, 0, stream>>>(t, mean, rsig, gm, bm,
                                                           abf, Nn * dh, dh);

        // ---- t = h1 @ W2 ----
        if (dp & 127)  // partial output tile: zero padded Bt so garbage cols stay finite
            k_zero_u16<<<cdiv(HID_ * HID_, B), B, 0, stream>>>(wbf, HID_ * HID_);
        k_transpose_bf16<<<cdiv(dh * dp, B), B, 0, stream>>>(W2, wbf, dh, dp);
        {
            dim3 grid(cdiv(Nn, 64), cdiv(dp, 128));
            k_gemm_bf16<<<grid, 128, 0, stream>>>(abf, wbf, t, Nn, dh, dp);
        }

        // ---- h = relu(BN(t)) ----
        k_zero_f32<<<cdiv(dp, B), B, 0, stream>>>(csum, dp);
        k_zero_f32<<<cdiv(dp, B), B, 0, stream>>>(csq, dp);
        k_col_stats<<<cdiv(Nn, 128), B, 0, stream>>>(t, csum, csq, Nn, dp, 128);
        k_finalize_stats<<<cdiv(dp, B), B, 0, stream>>>(csum, csq, mean, rsig, Nn, dp);
        k_bn_relu_f32<<<cdiv(Nn * dp, B), B, 0, stream>>>(t, mean, rsig, go, bo,
                                                          h, Nn * dp, dp);

        // ---- pool + prediction head i+1 ----
        k_zero_f32<<<cdiv(Gg * dp, B), B, 0, stream>>>(pooled, Gg * dp);
        k_pool<<<cdiv(Nn * dp, B), B, 0, stream>>>(h, gid, pooled, Nn, dp);
        k_score<<<1, Gg * OUT_, 0, stream>>>(pooled, (const float*)d_in[37 + i + 1],
                                             (const float*)d_in[44 + i + 1], score, dp);
    }

    k_logsoftmax<<<1, Gg, 0, stream>>>(score, (float*)d_out);
}
